// ST_BLOCK_5_47158740910292
// MI455X (gfx1250) — matrix operand order repaired and checked by reading, not compiled, not run on hardware
//
#include <hip/hip_runtime.h>

// ---------------------------------------------------------------------------
// ST-Block (Chebyshev graph-conv LSTM) for MI455X / gfx1250, wave32 + WMMA.
// B=32, C_IN=32, C_OUT=64, N=1024, T=24, K=3, C_MID=96.
// All GEMMs via v_wmma_f32_16x16x32_bf16 (f32 accumulate).
// ---------------------------------------------------------------------------

typedef __attribute__((ext_vector_type(16))) __bf16 v16bf;
typedef __attribute__((ext_vector_type(8)))  float  v8f;
typedef __attribute__((ext_vector_type(4)))  unsigned int uint4v;

union BF16x16 { v16bf v; uint4v u[2]; };
union BF16x8  { __bf16 h[8]; uint4v u; };

#define NN   1024
#define BB   32
#define CMID 96
#define COUT 64
#define TT   24
#define KI   288   // K*C_MID

// --- fragment loaders (match CDNA5 ISA §7.12.2 VGPR layouts) ----------------
// A 16x32 bf16: lane = row (L%16); halves: K in [hi*8, hi*8+8) and [16+hi*8, ...)
__device__ __forceinline__ v16bf load_a_frag(const __bf16* __restrict__ base,
                                             int ld, int row0, int k0, int lane) {
  const int lo = lane & 15, hi = lane >> 4;
  const __bf16* p = base + (size_t)(row0 + lo) * ld + k0 + hi * 8;
  BF16x16 r;
  r.u[0] = *reinterpret_cast<const uint4v*>(p);
  r.u[1] = *reinterpret_cast<const uint4v*>(p + 16);
  return r.v;
}
// B 32x16 bf16: lane = output column (L%16); halves: K in [hi*8, hi*8+8) and [16+hi*8, ...)
// (same as A). `base` is laid out [N_out][K] (K contiguous), ld = K-stride.
__device__ __forceinline__ v16bf load_b_frag(const __bf16* __restrict__ base,
                                             int ld, int col0, int k0, int lane) {
  const int lo = lane & 15, hi = lane >> 4;
  const __bf16* p = base + (size_t)(col0 + lo) * ld + k0 + hi * 8;
  BF16x16 r;
  r.u[0] = *reinterpret_cast<const uint4v*>(p);
  r.u[1] = *reinterpret_cast<const uint4v*>(p + 16);
  return r.v;
}

__device__ __forceinline__ v8f wmma_bf16(v16bf a, v16bf b, v8f c) {
  return __builtin_amdgcn_wmma_f32_16x16x32_bf16(false, a, false, b,
                                                 (short)0, c, false, false);
}

__device__ __forceinline__ uint4v pack_bf8(v8f a) {
  BF16x8 r;
#pragma unroll
  for (int e = 0; e < 8; ++e) r.h[e] = (__bf16)a[e];
  return r.u;
}

__device__ __forceinline__ float sigmoidf(float v) {
  return 1.0f / (1.0f + __expf(-v));
}

// --- setup: adj -> bf16 row-major (Lap1) + transposed copy ------------------
__global__ __launch_bounds__(256) void prep_adj(const float* __restrict__ adj,
                                                __bf16* __restrict__ lap0,
                                                __bf16* __restrict__ adjT) {
  __shared__ __bf16 s[64][72];
  const int r0 = blockIdx.y * 64, c0 = blockIdx.x * 64;
  for (int idx = threadIdx.x; idx < 4096; idx += 256) {
    const int rr = idx >> 6, cc = idx & 63;
    const __bf16 hv = (__bf16)adj[(size_t)(r0 + rr) * NN + c0 + cc];
    lap0[(size_t)(r0 + rr) * NN + c0 + cc] = hv;
    s[rr][cc] = hv;
  }
  __syncthreads();
  for (int idx = threadIdx.x; idx < 4096; idx += 256) {
    const int cc = idx >> 6, rr = idx & 63;
    adjT[(size_t)(c0 + cc) * NN + r0 + rr] = s[rr][cc];
  }
}

// --- setup: Lap2 = 2*adj@adj - I (WMMA, bf16 in / bf16 out) -----------------
__global__ __launch_bounds__(128) void cheb2(const __bf16* __restrict__ lap0,
                                             const __bf16* __restrict__ adjT,
                                             __bf16* __restrict__ lap1) {
  const int lane = threadIdx.x & 31;
  const int wave = threadIdx.x >> 5;
  const int wm = wave >> 1, wn = wave & 1;
  const int q0 = blockIdx.y * 128 + wm * 64;
  const int n0 = blockIdx.x * 128 + wn * 64;

  v8f acc[4][4];
#pragma unroll
  for (int i = 0; i < 4; ++i)
#pragma unroll
    for (int j = 0; j < 4; ++j) acc[i][j] = (v8f)0.0f;

  for (int k0 = 0; k0 < NN; k0 += 32) {
    v16bf a[4], b[4];
#pragma unroll
    for (int mt = 0; mt < 4; ++mt) a[mt] = load_a_frag(lap0, NN, q0 + mt * 16, k0, lane);
#pragma unroll
    for (int nt = 0; nt < 4; ++nt) b[nt] = load_b_frag(adjT, NN, n0 + nt * 16, k0, lane);
#pragma unroll
    for (int mt = 0; mt < 4; ++mt)
#pragma unroll
      for (int nt = 0; nt < 4; ++nt) acc[mt][nt] = wmma_bf16(a[mt], b[nt], acc[mt][nt]);
  }
  const int lo = lane & 15, hi = lane >> 4;
#pragma unroll
  for (int mt = 0; mt < 4; ++mt)
#pragma unroll
    for (int nt = 0; nt < 4; ++nt) {
      const int n = n0 + nt * 16 + lo;
#pragma unroll
      for (int jj = 0; jj < 8; ++jj) {
        const int q = q0 + mt * 16 + jj + 8 * hi;
        const float v = 2.0f * acc[mt][nt][jj] - (q == n ? 1.0f : 0.0f);
        lap1[(size_t)q * NN + n] = (__bf16)v;
      }
    }
}

// --- setup: permute W columns (i' = k*96 + c) and convert to bf16 -----------
__global__ __launch_bounds__(256) void prep_w(const float* __restrict__ W,
                                              __bf16* __restrict__ Wp) {
  const int i = blockIdx.x * 256 + threadIdx.x;
  if (i < 256 * KI) {
    const int o = i / KI, ip = i % KI;
    const int k = ip / CMID, c = ip % CMID;
    Wp[i] = (__bf16)W[o * KI + c * 3 + k];
  }
}

__global__ __launch_bounds__(256) void zero_f32(float* __restrict__ p, int n) {
  const int i = blockIdx.x * 256 + threadIdx.x;
  if (i < n) p[i] = 0.0f;
}

// --- per step: build tem (A of GEMM1) + identity (k=0) slice of xrT ---------
__global__ __launch_bounds__(256) void pack_tem(const float* __restrict__ x,
                                                const float* __restrict__ h_state,
                                                __bf16* __restrict__ tem,
                                                __bf16* __restrict__ xrT,
                                                int t) {
  __shared__ __bf16 s[CMID][72];
  const int b = blockIdx.y;
  const int n0 = blockIdx.x * 64;
  for (int idx = threadIdx.x; idx < CMID * 64; idx += 256) {
    const int c = idx >> 6;
    const int nn = idx & 63;
    const int n = n0 + nn;
    float v;
    if (c < 32) v = x[(((size_t)(b * 32 + c)) * NN + n) * TT + t];
    else        v = h_state[((size_t)(b * COUT + (c - 32))) * NN + n];
    const __bf16 hv = (__bf16)v;
    tem[((size_t)(b * CMID + c)) * NN + n] = hv;
    s[c][nn] = hv;
  }
  __syncthreads();
  for (int idx = threadIdx.x; idx < 64 * 12; idx += 256) {
    const int nn = idx / 12;
    const int ch = (idx % 12) * 8;
    BF16x8 r;
#pragma unroll
    for (int e = 0; e < 8; ++e) r.h[e] = s[ch + e][nn];
    *reinterpret_cast<uint4v*>(xrT + ((size_t)(b * NN + n0 + nn)) * KI + ch) = r.u;
  }
}

// --- per step GEMM1: xrT[b][q][k*96+c] = sum_n tem[b,c,n]*Lap_k[q,n] --------
__global__ __launch_bounds__(128) void gemm1_cheb(const __bf16* __restrict__ tem,
                                                  const __bf16* __restrict__ lap,
                                                  __bf16* __restrict__ xrT) {
  const int lane = threadIdx.x & 31;
  const int wave = threadIdx.x >> 5;
  const int wm = wave >> 1, wn = wave & 1;
  const int kIdx = blockIdx.z;                         // 0 -> L1, 1 -> L2
  const int r0 = blockIdx.y * 128 + wm * 64;           // row = b*96+c
  const int q0 = blockIdx.x * 128 + wn * 64;
  const __bf16* lapk = lap + (size_t)kIdx * NN * NN;   // [q][n] row-major

  v8f acc[4][4];
#pragma unroll
  for (int i = 0; i < 4; ++i)
#pragma unroll
    for (int j = 0; j < 4; ++j) acc[i][j] = (v8f)0.0f;

  for (int k0 = 0; k0 < NN; k0 += 32) {
    v16bf a[4], b[4];
#pragma unroll
    for (int mt = 0; mt < 4; ++mt) a[mt] = load_a_frag(tem, NN, r0 + mt * 16, k0, lane);
#pragma unroll
    for (int nt = 0; nt < 4; ++nt) b[nt] = load_b_frag(lapk, NN, q0 + nt * 16, k0, lane);
#pragma unroll
    for (int mt = 0; mt < 4; ++mt)
#pragma unroll
      for (int nt = 0; nt < 4; ++nt) acc[mt][nt] = wmma_bf16(a[mt], b[nt], acc[mt][nt]);
  }

  const int lo = lane & 15, hi = lane >> 4;
#pragma unroll
  for (int mt = 0; mt < 4; ++mt) {
    const int grow = r0 + mt * 16;              // 16-row tile never straddles b (96%16==0)
    const int bb = grow / CMID, c0 = grow % CMID;
#pragma unroll
    for (int nt = 0; nt < 4; ++nt) {
      const int q = q0 + nt * 16 + lo;
      const uint4v pk = pack_bf8(acc[mt][nt]);  // 8 consecutive c, one 16B store
      __bf16* dst = xrT + ((size_t)(bb * NN + q)) * KI + (kIdx + 1) * CMID + c0 + hi * 8;
      *reinterpret_cast<uint4v*>(dst) = pk;
    }
  }
}

// --- per step GEMM2 + fused LSTM gates --------------------------------------
// Wave owns M-tiles {ch0, ch0+64, ch0+128, ch0+192}: all 4 gates of the same
// 16 cells land in the same lane/VGPR slots -> gate math directly on accums.
__global__ __launch_bounds__(256) void gemm2_gates(const __bf16* __restrict__ Wp,
                                                   const __bf16* __restrict__ xrT,
                                                   const float* __restrict__ bias,
                                                   float* __restrict__ c_state,
                                                   float* __restrict__ h_state,
                                                   float* __restrict__ out,
                                                   int t) {
  const int lane = threadIdx.x & 31;
  const int wave = threadIdx.x >> 5;     // 0..7
  const int wm = wave >> 1;              // 0..3 -> cells [wm*16, wm*16+16)
  const int wn = wave & 1;
  const int b = blockIdx.y;
  const int ch0 = wm * 16;
  const int n0 = blockIdx.x * 128 + wn * 64;
  const __bf16* Bm = xrT + (size_t)b * NN * KI;   // [n][i'] (i' contiguous)

  v8f acc[4][4];                         // [gate][n-tile]
#pragma unroll
  for (int i = 0; i < 4; ++i)
#pragma unroll
    for (int j = 0; j < 4; ++j) acc[i][j] = (v8f)0.0f;

  for (int k0 = 0; k0 < KI; k0 += 32) {  // 9 chunks
    v16bf a[4], bb[4];
#pragma unroll
    for (int g = 0; g < 4; ++g) a[g] = load_a_frag(Wp, KI, ch0 + g * 64, k0, lane);
#pragma unroll
    for (int nt = 0; nt < 4; ++nt) bb[nt] = load_b_frag(Bm, KI, n0 + nt * 16, k0, lane);
#pragma unroll
    for (int g = 0; g < 4; ++g)
#pragma unroll
      for (int nt = 0; nt < 4; ++nt) acc[g][nt] = wmma_bf16(a[g], bb[nt], acc[g][nt]);
  }

  const int lo = lane & 15, hi = lane >> 4;
  float bi[4][8];
#pragma unroll
  for (int g = 0; g < 4; ++g) {
    const float* bp = bias + ch0 + g * 64 + hi * 8;
#pragma unroll
    for (int jj = 0; jj < 8; ++jj) bi[g][jj] = bp[jj];
  }

#pragma unroll
  for (int nt = 0; nt < 4; ++nt) {
    const int n = n0 + nt * 16 + lo;
#pragma unroll
    for (int jj = 0; jj < 8; ++jj) {
      const int cell = ch0 + jj + 8 * hi;
      const size_t idx = ((size_t)(b * COUT + cell)) * NN + n;
      const float iv = acc[0][nt][jj] + bi[0][jj];
      const float jv = acc[1][nt][jj] + bi[1][jj];
      const float fv = acc[2][nt][jj] + bi[2][jj];
      const float ov = acc[3][nt][jj] + bi[3][jj];
      const float cold = c_state[idx];
      const float cnew = cold * sigmoidf(fv) + sigmoidf(iv) * tanhf(jv);
      const float hnew = tanhf(cnew) * sigmoidf(ov);
      c_state[idx] = cnew;
      h_state[idx] = hnew;
      out[idx * TT + t] = hnew;
    }
  }
}

// ---------------------------------------------------------------------------
extern "C" void kernel_launch(void* const* d_in, const int* in_sizes, int n_in,
                              void* d_out, int out_size, void* d_ws, size_t ws_size,
                              hipStream_t stream) {
  const float* x    = (const float*)d_in[0];  // [32][32][1024][24]
  const float* adj  = (const float*)d_in[1];  // [1024][1024]
  const float* W    = (const float*)d_in[2];  // [256][288]
  const float* bias = (const float*)d_in[3];  // [256]
  float* out = (float*)d_out;                 // [32][64][1024][24]

  char* ws = (char*)d_ws;
  size_t off = 0;
  __bf16* lap = (__bf16*)(ws + off);   off += (size_t)2 * NN * NN * 2;     // L1,L2 bf16 [q][n]
  __bf16* adjT = (__bf16*)(ws + off);  off += (size_t)NN * NN * 2;         // adj^T bf16
  __bf16* Wp = (__bf16*)(ws + off);    off += (size_t)256 * KI * 2;        // permuted W bf16
  off = (off + 255) & ~(size_t)255;
  __bf16* tem = (__bf16*)(ws + off);   off += (size_t)BB * CMID * NN * 2;  // [3072][1024]
  __bf16* xrT = (__bf16*)(ws + off);   off += (size_t)BB * NN * KI * 2;    // [32][1024][288]
  float* h_state = (float*)(ws + off); off += (size_t)BB * COUT * NN * 4;
  float* c_state = (float*)(ws + off); off += (size_t)BB * COUT * NN * 4;
  (void)ws_size; (void)in_sizes; (void)n_in; (void)out_size;

  // ---- setup (graph-capturable, deterministic, re-done every call) ----
  prep_adj<<<dim3(16, 16), 256, 0, stream>>>(adj, lap, adjT);
  cheb2<<<dim3(8, 8), 128, 0, stream>>>(lap, adjT, lap + (size_t)NN * NN);
  prep_w<<<dim3((256 * KI + 255) / 256), 256, 0, stream>>>(W, Wp);
  const int nState = 2 * BB * COUT * NN;  // h then c, contiguous
  zero_f32<<<dim3((nState + 255) / 256), 256, 0, stream>>>(h_state, nState);

  // ---- sequential recurrence over T (stream-ordered) ----
  for (int t = 0; t < TT; ++t) {
    pack_tem<<<dim3(16, BB), 256, 0, stream>>>(x, h_state, tem, xrT, t);
    gemm1_cheb<<<dim3(8, 24, 2), 128, 0, stream>>>(tem, lap, xrT);
    gemm2_gates<<<dim3(8, BB), 256, 0, stream>>>(Wp, xrT, bias, c_state, h_state, out, t);
  }
}
